// ImprovedCGCNN_49460843381244
// MI455X (gfx1250) — compile-verified, matrix operand order checked by reading, not run on hardware
//
#include <hip/hip_runtime.h>

typedef __attribute__((ext_vector_type(16))) _Float16 v16h;
typedef __attribute__((ext_vector_type(8)))  float    v8f;
typedef __attribute__((ext_vector_type(8)))  unsigned int v8u;

#define N_NODES 50000
#define N_EDGES 800000
#define N_GRAPHS 1024
#define HDIM 128
#define NLAYER 5
#define N_TILES (N_EDGES / 16)

__device__ __forceinline__ float silu_f(float v) {
    return v / (1.0f + __expf(-v));
}

// Build one 16-half WMMA fragment (A or B layout) from a row of fp16:
// element r covers k = (r>>2)*16 + hi*8 + (r&3)*2  -> two contiguous 16B loads.
__device__ __forceinline__ v16h load_frag(const _Float16* row, int kbase, int hi) {
    uint4 lo = *(const uint4*)(row + kbase + hi * 8);
    uint4 hw = *(const uint4*)(row + kbase + 16 + hi * 8);
    v8u au;
    au[0] = lo.x; au[1] = lo.y; au[2] = lo.z; au[3] = lo.w;
    au[4] = hw.x; au[5] = hw.y; au[6] = hw.z; au[7] = hw.w;
    return __builtin_bit_cast(v16h, au);
}

// ---------------------------------------------------------------------------
// Utility: zero a float buffer (grid-stride)
// ---------------------------------------------------------------------------
__global__ void zero_f32(float* __restrict__ p, long n) {
    long i = (long)blockIdx.x * blockDim.x + threadIdx.x;
    long st = (long)gridDim.x * blockDim.x;
    for (; i < n; i += st) p[i] = 0.0f;
}

// ---------------------------------------------------------------------------
// Weight prep: transpose + fp32->fp16.  w1t[l][n][k] = w1[l][k][n]
// ---------------------------------------------------------------------------
__global__ void prep_w1(const float* __restrict__ w, _Float16* __restrict__ wt) {
    long total = (long)NLAYER * 256 * 256;
    long i = (long)blockIdx.x * blockDim.x + threadIdx.x;
    long st = (long)gridDim.x * blockDim.x;
    for (; i < total; i += st) {
        long l = i / 65536, rem = i % 65536;
        long n = rem / 256, k = rem % 256;
        wt[i] = (_Float16)w[l * 65536 + k * 256 + n];
    }
}

__global__ void prep_w2(const float* __restrict__ w, _Float16* __restrict__ wt) {
    long total = (long)NLAYER * 128 * 256;
    long i = (long)blockIdx.x * blockDim.x + threadIdx.x;
    long st = (long)gridDim.x * blockDim.x;
    for (; i < total; i += st) {
        long l = i / 32768, rem = i % 32768;
        long n = rem / 256, k = rem % 256;          // n<128 (out), k<256 (in)
        wt[i] = (_Float16)w[l * 32768 + k * 128 + n];
    }
}

// ---------------------------------------------------------------------------
// Node embedding: x = silu(LN(atom @ emb_w + emb_b)); one block (128 thr)/node
// ---------------------------------------------------------------------------
__global__ __launch_bounds__(128) void node_embed(
    const float* __restrict__ atom, const float* __restrict__ ew,
    const float* __restrict__ eb, const float* __restrict__ g,
    const float* __restrict__ b, float* __restrict__ x,
    _Float16* __restrict__ xh)
{
    int n = blockIdx.x, j = threadIdx.x;
    float a0 = atom[n * 4 + 0], a1 = atom[n * 4 + 1];
    float a2 = atom[n * 4 + 2], a3 = atom[n * 4 + 3];
    float y = a0 * ew[0 * HDIM + j] + a1 * ew[1 * HDIM + j] +
              a2 * ew[2 * HDIM + j] + a3 * ew[3 * HDIM + j] + eb[j];
    __shared__ float s1[128], s2[128];
    s1[j] = y; s2[j] = y * y;
    __syncthreads();
    for (int off = 64; off > 0; off >>= 1) {
        if (j < off) { s1[j] += s1[j + off]; s2[j] += s2[j + off]; }
        __syncthreads();
    }
    float mean = s1[0] * (1.0f / HDIM);
    float var  = s2[0] * (1.0f / HDIM) - mean * mean;
    float rstd = rsqrtf(var + 1e-5f);
    float v = silu_f((y - mean) * rstd * g[j] + b[j]);
    x[(long)n * HDIM + j]  = v;
    xh[(long)n * HDIM + j] = (_Float16)v;
}

// ---------------------------------------------------------------------------
// Edge embedding: e_h = fp16(silu(nbr_fea @ edge_w + edge_b)); 1 block/edge
// ---------------------------------------------------------------------------
__global__ __launch_bounds__(128) void edge_embed(
    const float* __restrict__ nbr, const float* __restrict__ ew,
    const float* __restrict__ eb, _Float16* __restrict__ eh)
{
    long e = blockIdx.x; int j = threadIdx.x;
    __shared__ float row[41];
    if (j < 41) row[j] = nbr[e * 41 + j];
    __syncthreads();
    float acc = eb[j];
    #pragma unroll
    for (int i = 0; i < 41; ++i) acc += row[i] * ew[i * HDIM + j];
    eh[e * HDIM + j] = (_Float16)silu_f(acc);
}

// ---------------------------------------------------------------------------
// Degree count (dst)
// ---------------------------------------------------------------------------
__global__ void count_deg(const int* __restrict__ nbr_idx, float* __restrict__ cnt) {
    long i = (long)blockIdx.x * blockDim.x + threadIdx.x;
    long st = (long)gridDim.x * blockDim.x;
    for (; i < N_EDGES; i += st)
        atomicAdd(&cnt[nbr_idx[i * 2 + 1]], 1.0f);
}

// ---------------------------------------------------------------------------
// Core: persistent-block edge message MLP via WMMA, M-blocked x2.
//  - Stage layer weights (W1 128KB + W2 64KB, fp16, [n][k]) into LDS once.
//  - 8 waves/block; each wave processes TWO 16-edge tiles per iteration so
//    every B-fragment ds_load_b128 feeds two v_wmma ops (LDS:MMA ratio 1:1).
//  LDS: 128K (W1) + 64K (W2) + 8*16K (hidden) = 320KB (CDNA5 WGP max).
// ---------------------------------------------------------------------------
__global__ __launch_bounds__(256) void edge_conv_wmma(
    const _Float16* __restrict__ xh, const _Float16* __restrict__ eh,
    const int* __restrict__ nbr_idx,
    const _Float16* __restrict__ w1t, const float* __restrict__ b1,
    const _Float16* __restrict__ w2t, const float* __restrict__ b2,
    float* __restrict__ aggr)
{
    __shared__ _Float16 w1s[256 * 256];         // 128 KB
    __shared__ _Float16 w2s[128 * 256];         //  64 KB
    __shared__ _Float16 hid[8 * 2 * 16 * 256];  // 128 KB (16KB per wave)

    const int tid  = threadIdx.x;
    const int wave = tid >> 5;
    const int lane = tid & 31;
    const int hi   = lane >> 4;
    const int ln15 = lane & 15;

    // ---- cooperative weight staging (b128 copies) ----
    {
        const uint4* s1 = (const uint4*)w1t;
        uint4*       d1 = (uint4*)w1s;
        for (int i = tid; i < 256 * 256 / 8; i += 256) d1[i] = s1[i];
        const uint4* s2 = (const uint4*)w2t;
        uint4*       d2 = (uint4*)w2s;
        for (int i = tid; i < 128 * 256 / 8; i += 256) d2[i] = s2[i];
    }
    __syncthreads();

    _Float16* hid0 = &hid[wave * 2 * 16 * 256];
    _Float16* hid1 = hid0 + 16 * 256;

    // Hoisted per-lane biases (lane-dependent only)
    float b1v[16], b2v[8];
    #pragma unroll
    for (int nt = 0; nt < 16; ++nt) b1v[nt] = b1[nt * 16 + ln15];
    #pragma unroll
    for (int nt = 0; nt < 8; ++nt)  b2v[nt] = b2[nt * 16 + ln15];

    const long wave_id = (long)blockIdx.x * 8 + wave;
    const long n_waves = (long)gridDim.x * 8;
    const long n_pairs = N_TILES / 2;           // 50000/2 = 25000 exactly

    for (long p = wave_id; p < n_pairs; p += n_waves) {
        const long e0 = (p * 2) * 16;
        const long e1 = e0 + 16;

        // A-fragment row pointers (row m = ln15) for both tiles
        const int s0 = nbr_idx[(e0 + ln15) * 2 + 0];
        const int s1 = nbr_idx[(e1 + ln15) * 2 + 0];
        const _Float16* xrow0 = xh + (long)s0 * HDIM;
        const _Float16* xrow1 = xh + (long)s1 * HDIM;
        const _Float16* erow0 = eh + (e0 + ln15) * HDIM;
        const _Float16* erow1 = eh + (e1 + ln15) * HDIM;

        // dst indices for rows m = hi*8 + r (matches C/D layout)
        int dst0[8], dst1[8];
        #pragma unroll
        for (int r = 0; r < 8; ++r) {
            dst0[r] = nbr_idx[(e0 + hi * 8 + r) * 2 + 1];
            dst1[r] = nbr_idx[(e1 + hi * 8 + r) * 2 + 1];
        }

        // ---- load A fragments for both tiles (8 K-tiles of 32 each) ----
        v16h a0[8], a1[8];
        #pragma unroll
        for (int kt = 0; kt < 8; ++kt) {
            const int kb = (kt & 3) * 32;
            a0[kt] = load_frag((kt < 4) ? xrow0 : erow0, kb, hi);
            a1[kt] = load_frag((kt < 4) ? xrow1 : erow1, kb, hi);
        }

        // ---- GEMM1: (2x)16x256 @ 256x256 -> silu -> hidden (LDS) ----
        for (int nt = 0; nt < 16; ++nt) {
            v8f c0 = {}, c1 = {};
            const _Float16* wb = w1s + (nt * 16 + ln15) * 256;
            #pragma unroll
            for (int kt = 0; kt < 8; ++kt) {
                v16h bf = load_frag(wb, kt * 32, hi);
                c0 = __builtin_amdgcn_wmma_f32_16x16x32_f16(
                         false, a0[kt], false, bf, (short)0, c0, false, false);
                c1 = __builtin_amdgcn_wmma_f32_16x16x32_f16(
                         false, a1[kt], false, bf, (short)0, c1, false, false);
            }
            #pragma unroll
            for (int r = 0; r < 8; ++r) {
                const int idx = (hi * 8 + r) * 256 + nt * 16 + ln15;
                hid0[idx] = (_Float16)silu_f(c0[r] + b1v[nt]);
                hid1[idx] = (_Float16)silu_f(c1[r] + b1v[nt]);
            }
        }

        // ---- reload hidden as A fragments (wave-private LDS, in-order) ----
        v16h h0[8], h1[8];
        #pragma unroll
        for (int kt = 0; kt < 8; ++kt) {
            h0[kt] = load_frag(hid0 + ln15 * 256, kt * 32, hi);
            h1[kt] = load_frag(hid1 + ln15 * 256, kt * 32, hi);
        }

        // ---- GEMM2: (2x)16x256 @ 256x128 -> silu -> atomic scatter ----
        for (int nt = 0; nt < 8; ++nt) {
            v8f c0 = {}, c1 = {};
            const _Float16* wb = w2s + (nt * 16 + ln15) * 256;
            #pragma unroll
            for (int kt = 0; kt < 8; ++kt) {
                v16h bf = load_frag(wb, kt * 32, hi);
                c0 = __builtin_amdgcn_wmma_f32_16x16x32_f16(
                         false, h0[kt], false, bf, (short)0, c0, false, false);
                c1 = __builtin_amdgcn_wmma_f32_16x16x32_f16(
                         false, h1[kt], false, bf, (short)0, c1, false, false);
            }
            #pragma unroll
            for (int r = 0; r < 8; ++r) {
                const int col = nt * 16 + ln15;
                atomicAdd(&aggr[(long)dst0[r] * HDIM + col],
                          silu_f(c0[r] + b2v[nt]));
                atomicAdd(&aggr[(long)dst1[r] * HDIM + col],
                          silu_f(c1[r] + b2v[nt]));
            }
        }
    }
}

// ---------------------------------------------------------------------------
// Layer update: x = LN(x + aggr/cnt); rewrite x (fp32) and x_h (fp16)
// ---------------------------------------------------------------------------
__global__ __launch_bounds__(128) void node_update(
    float* __restrict__ x, _Float16* __restrict__ xh,
    const float* __restrict__ aggr, const float* __restrict__ cnt,
    const float* __restrict__ g, const float* __restrict__ b)
{
    int n = blockIdx.x, j = threadIdx.x;
    float c = cnt[n]; if (c < 1.0f) c = 1.0f;
    float y = x[(long)n * HDIM + j] + aggr[(long)n * HDIM + j] / c;
    __shared__ float s1[128], s2[128];
    s1[j] = y; s2[j] = y * y;
    __syncthreads();
    for (int off = 64; off > 0; off >>= 1) {
        if (j < off) { s1[j] += s1[j + off]; s2[j] += s2[j + off]; }
        __syncthreads();
    }
    float mean = s1[0] * (1.0f / HDIM);
    float var  = s2[0] * (1.0f / HDIM) - mean * mean;
    float rstd = rsqrtf(var + 1e-5f);
    float v = (y - mean) * rstd * g[j] + b[j];
    x[(long)n * HDIM + j]  = v;
    xh[(long)n * HDIM + j] = (_Float16)v;
}

// ---------------------------------------------------------------------------
// Graph mean-pool (atomics)
// ---------------------------------------------------------------------------
__global__ __launch_bounds__(128) void pool(
    const float* __restrict__ x, const int* __restrict__ bmap,
    float* __restrict__ cry, float* __restrict__ gcnt)
{
    int n = blockIdx.x, j = threadIdx.x;
    int g = bmap[n];
    atomicAdd(&cry[(long)g * HDIM + j], x[(long)n * HDIM + j]);
    if (j == 0) atomicAdd(&gcnt[g], 1.0f);
}

// ---------------------------------------------------------------------------
// Head MLP: 128 -> 128 -> 64 -> 3 per graph (one block / graph)
// ---------------------------------------------------------------------------
__global__ __launch_bounds__(128) void head_mlp(
    const float* __restrict__ cry, const float* __restrict__ gcnt,
    const float* __restrict__ w1, const float* __restrict__ b1,
    const float* __restrict__ w2, const float* __restrict__ b2,
    const float* __restrict__ w3, const float* __restrict__ b3,
    float* __restrict__ out)
{
    int g = blockIdx.x, j = threadIdx.x;
    __shared__ float cr[128], h1[128], h2[64];
    float gc = gcnt[g]; if (gc < 1.0f) gc = 1.0f;
    cr[j] = cry[(long)g * HDIM + j] / gc;
    __syncthreads();
    float acc = b1[j];
    #pragma unroll 8
    for (int k = 0; k < 128; ++k) acc += cr[k] * w1[k * 128 + j];
    h1[j] = silu_f(acc);
    __syncthreads();
    if (j < 64) {
        acc = b2[j];
        #pragma unroll 8
        for (int k = 0; k < 128; ++k) acc += h1[k] * w2[k * 64 + j];
        h2[j] = silu_f(acc);
    }
    __syncthreads();
    if (j < 3) {
        acc = b3[j];
        #pragma unroll 8
        for (int k = 0; k < 64; ++k) acc += h2[k] * w3[k * 3 + j];
        out[(long)g * 3 + j] = acc;
    }
}

// ---------------------------------------------------------------------------
extern "C" void kernel_launch(void* const* d_in, const int* in_sizes, int n_in,
                              void* d_out, int out_size, void* d_ws, size_t ws_size,
                              hipStream_t stream) {
    const float* atom_fea = (const float*)d_in[0];
    const float* nbr_fea  = (const float*)d_in[1];
    const int*   nbr_idx  = (const int*)d_in[2];
    const int*   bmap     = (const int*)d_in[3];
    const float* emb_w    = (const float*)d_in[4];
    const float* emb_b    = (const float*)d_in[5];
    const float* emb_ln_g = (const float*)d_in[6];
    const float* emb_ln_b = (const float*)d_in[7];
    const float* edge_w   = (const float*)d_in[8];
    const float* edge_b   = (const float*)d_in[9];
    const float* conv_w1  = (const float*)d_in[10];
    const float* conv_b1  = (const float*)d_in[11];
    const float* conv_w2  = (const float*)d_in[12];
    const float* conv_b2  = (const float*)d_in[13];
    const float* ln_g     = (const float*)d_in[14];
    const float* ln_b     = (const float*)d_in[15];
    const float* out_w1   = (const float*)d_in[16];
    const float* out_b1   = (const float*)d_in[17];
    const float* out_w2   = (const float*)d_in[18];
    const float* out_b2   = (const float*)d_in[19];
    const float* out_w3   = (const float*)d_in[20];
    const float* out_b3   = (const float*)d_in[21];

    // workspace carve-up (256-byte aligned)
    char* ws = (char*)d_ws;
    size_t off = 0;
    auto carve = [&](size_t bytes) {
        char* p = ws + off;
        off = (off + bytes + 255) & ~(size_t)255;
        return p;
    };
    float*    X    = (float*)   carve((size_t)N_NODES * HDIM * 4);
    _Float16* XH   = (_Float16*)carve((size_t)N_NODES * HDIM * 2);
    _Float16* EH   = (_Float16*)carve((size_t)N_EDGES * HDIM * 2);
    float*    AGGR = (float*)   carve((size_t)N_NODES * HDIM * 4);
    float*    CNT  = (float*)   carve((size_t)N_NODES * 4);
    _Float16* W1T  = (_Float16*)carve((size_t)NLAYER * 256 * 256 * 2);
    _Float16* W2T  = (_Float16*)carve((size_t)NLAYER * 128 * 256 * 2);
    float*    CRY  = (float*)   carve((size_t)N_GRAPHS * HDIM * 4);
    float*    GCNT = (float*)   carve((size_t)N_GRAPHS * 4);
    (void)ws_size; (void)in_sizes; (void)n_in; (void)out_size;

    // weight prep (fp32 -> transposed fp16)
    prep_w1<<<640, 256, 0, stream>>>(conv_w1, W1T);
    prep_w2<<<640, 256, 0, stream>>>(conv_w2, W2T);

    // embeddings
    node_embed<<<N_NODES, 128, 0, stream>>>(atom_fea, emb_w, emb_b,
                                            emb_ln_g, emb_ln_b, X, XH);
    edge_embed<<<N_EDGES, 128, 0, stream>>>(nbr_fea, edge_w, edge_b, EH);

    // degree counts
    zero_f32<<<256, 256, 0, stream>>>(CNT, (long)N_NODES);
    count_deg<<<2048, 256, 0, stream>>>(nbr_idx, CNT);

    // message-passing layers
    for (int l = 0; l < NLAYER; ++l) {
        zero_f32<<<4096, 256, 0, stream>>>(AGGR, (long)N_NODES * HDIM);
        edge_conv_wmma<<<512, 256, 0, stream>>>(
            XH, EH, nbr_idx,
            W1T + (size_t)l * 256 * 256, conv_b1 + (size_t)l * 256,
            W2T + (size_t)l * 128 * 256, conv_b2 + (size_t)l * 128,
            AGGR);
        node_update<<<N_NODES, 128, 0, stream>>>(
            X, XH, AGGR, CNT, ln_g + (size_t)l * HDIM, ln_b + (size_t)l * HDIM);
    }

    // pooling + head
    zero_f32<<<512, 256, 0, stream>>>(CRY, (long)N_GRAPHS * HDIM);
    zero_f32<<<8, 256, 0, stream>>>(GCNT, (long)N_GRAPHS);
    pool<<<N_NODES, 128, 0, stream>>>(X, bmap, CRY, GCNT);
    head_mlp<<<N_GRAPHS, 128, 0, stream>>>(CRY, GCNT, out_w1, out_b1,
                                           out_w2, out_b2, out_w3, out_b3,
                                           (float*)d_out);
}
